// SlayerLoihiMM_27857157882558
// MI455X (gfx1250) — matrix operand order repaired and checked by reading, not compiled, not run on hardware
//
#include <hip/hip_runtime.h>
#include <hip/hip_bf16.h>
#include <math.h>

// ---------------------------------------------------------------------------
// Loihi SNN pipeline for MI455X (gfx1250, wave32, WMMA).
//   stage1: Y = W_tact x tact   (f16 WMMA, f32 acc) -> loihi scan -> f16 spikes
//   stage2: Y = W_vis  x vis    (dominant GEMM)     -> loihi scan -> f16 spikes
//   stage3: Y = W_combi x spikes (f16 B from ws)    -> loihi scan -> f32 out
// GEMM: MTx5 register tile per wave; BOTH A and B globally software-pipelined
// one K-chunk ahead; B coop-load = 1 row x 10 contiguous t per thread (one
// clamped address per iter, b64 clauses, packed ds_store_b32); validity select
// + cvt deferred to the LDS-store site so no wait lands next to the loads.
// ---------------------------------------------------------------------------

typedef _Float16 half_t;
typedef __attribute__((ext_vector_type(16))) _Float16 v16h;
typedef __attribute__((ext_vector_type(8)))  _Float16 v8h;
typedef __attribute__((ext_vector_type(2)))  _Float16 v2h;
typedef __attribute__((ext_vector_type(8)))  float    v8f;
typedef __attribute__((ext_vector_type(2)))  float    v2f;

#define TT 400  // time steps: 5 grid-y blocks x (NT=5) tiles x 16

template <typename T> struct Vec2;
template <> struct Vec2<float>    { using type = v2f; };
template <> struct Vec2<_Float16> { using type = v2h; };

__device__ __forceinline__ v16h cat8(v8h lo, v8h hi) {
    return __builtin_shufflevector(lo, hi,
        0, 1, 2, 3, 4, 5, 6, 7, 8, 9, 10, 11, 12, 13, 14, 15);
}

// --- weight convert + zero-pad K -> Kp (multiple of 32) --------------------
__global__ void cvt_pad_weights(const float* __restrict__ W, half_t* __restrict__ Wh,
                                int M, int K, int Kp) {
    int i = blockIdx.x * blockDim.x + threadIdx.x;
    if (i >= M * Kp) return;
    int r = i / Kp, c = i - r * Kp;
    Wh[i] = (c < K) ? (half_t)W[(size_t)r * K + c] : (half_t)0.0f;
}

// --- batched GEMM: Y[n,m,t] = sum_k Wh[m,k] * X[n,k,t] ---------------------
// Block: 256 threads = 8 wave32. Block tile: (8*MT*16) M-rows x 80 T-cols.
template <typename TB, int MT>
__global__ __launch_bounds__(256)
void gemm_wmma(const half_t* __restrict__ Wh,  // [M, Kp] f16, zero-padded
               const TB* __restrict__ X,       // [NB, K, TT]
               float* __restrict__ Y,          // [NB, M, TT]
               int M, int K, int Kp) {
    using T2 = typename Vec2<TB>::type;
    constexpr int NT    = 5;           // 16-wide T tiles per wave
    constexpr int NTILE = NT * 16;     // 80 T columns per block
    constexpr int LDSS  = NTILE + 8;   // LDS row stride (halves): 176B, 16B-aligned

    __shared__ half_t Bs[32 * LDSS];

    const int tid  = threadIdx.x;
    const int lane = tid & 31;
    const int wave = tid >> 5;
    const int g    = lane >> 4;        // A K-group / C row-half select
    const int mr   = lane & 15;
    const int t0   = blockIdx.y * NTILE;
    const int n    = blockIdx.z;
    const int mblk = blockIdx.x * (8 * MT * 16) + wave * (MT * 16);

    // B coop load: thread owns K-row (tid&31) and 10 consecutive t columns
    const int k_l   = tid & 31;
    const int tbase = (tid >> 5) * 10;

    const TB* __restrict__ Xp = X + (size_t)n * K * TT + t0 + tbase;
    const half_t* __restrict__ Wbase = Wh + (size_t)(mblk + mr) * Kp + 8 * g;

    // ---- prologue: B chunk kb=0 (rows 0..31 always < K) ----
    T2 xrv[5];
    bool okr = true;
#pragma unroll
    for (int p = 0; p < 5; ++p)
        xrv[p] = *(const T2*)(Xp + (size_t)k_l * TT + 2 * p);

    // ---- prologue: A fragments for kb=0 ----
    v16h af[MT];
#pragma unroll
    for (int m = 0; m < MT; ++m) {
        const half_t* wa = Wbase + (size_t)(m * 16) * Kp;
        af[m] = cat8(*(const v8h*)(wa), *(const v8h*)(wa + 16));
    }

    v8f acc[MT * NT] = {};

    for (int kb = 0; kb < Kp; kb += 32) {
        __syncthreads();               // WAR vs previous frag reads
        // deferred select + convert + packed LDS store (b32 each)
#pragma unroll
        for (int p = 0; p < 5; ++p) {
            v2h pr;
            pr[0] = okr ? (half_t)xrv[p][0] : (half_t)0.0f;
            pr[1] = okr ? (half_t)xrv[p][1] : (half_t)0.0f;
            *(v2h*)&Bs[k_l * LDSS + tbase + 2 * p] = pr;
        }
        __syncthreads();               // RAW before frag reads

        // ---- prefetch next B chunk (one clamped row address) ----
        {
            const int krow = kb + 32 + k_l;
            const int kcl  = (krow < K) ? krow : (K - 1);
            const TB* rp = Xp + (size_t)kcl * TT;
#pragma unroll
            for (int p = 0; p < 5; ++p)
                xrv[p] = *(const T2*)(rp + 2 * p);
            okr = krow < K;
        }

        // ---- prefetch next A fragments (padded buffer: always in-bounds) ----
        v16h ac[MT];
        {
            const int kbn = (kb + 32 < Kp) ? (kb + 32) : kb;
#pragma unroll
            for (int m = 0; m < MT; ++m) {
                ac[m] = af[m];
                const half_t* wa = Wbase + (size_t)(m * 16) * Kp + kbn;
                af[m] = cat8(*(const v8h*)(wa), *(const v8h*)(wa + 16));
            }
        }

        // ---- B fragments from LDS: lane holds N = mr, element s -> K = 8g+s (s<8), 16+8g+s-8 (s>=8) ----
        v16h bf[NT];
#pragma unroll
        for (int nt = 0; nt < NT; ++nt) {
            const half_t* bp = Bs + nt * 16 + mr;
#pragma unroll
            for (int s = 0; s < 16; ++s)
                bf[nt][s] = bp[(s < 8 ? 8 * g + s : 8 + 8 * g + s) * LDSS];
        }

        // ---- WMMA on current fragments ----
#pragma unroll
        for (int m = 0; m < MT; ++m)
#pragma unroll
            for (int nt = 0; nt < NT; ++nt)
                acc[m * NT + nt] = __builtin_amdgcn_wmma_f32_16x16x32_f16(
                    false, ac[m], false, bf[nt], (short)0, acc[m * NT + nt],
                    false, false);
    }

    // ---- C write: VGPR r -> M = tile + r + 8*g, N = mr ----
#pragma unroll
    for (int m = 0; m < MT; ++m) {
#pragma unroll
        for (int nt = 0; nt < NT; ++nt) {
            float* p = Y + ((size_t)n * M + mblk + m * 16 + 8 * g) * TT
                         + t0 + nt * 16 + mr;
#pragma unroll
            for (int r = 0; r < 8; ++r)
                p[(size_t)r * TT] = acc[m * NT + nt][r];
        }
    }
}

// --- Loihi neuron scan + fused delay shift ---------------------------------
// iDecay=1024 -> u *= 0.75 (trunc); vDecay=128 -> v *= 0.96875 (trunc)
// theta = 80<<6 = 5120 ; wScale = 64. One thread per (n, o) neuron.
template <typename TOUT>
__global__ __launch_bounds__(256)
void loihi_scan(const float* __restrict__ Y,  // [NB, O, TT]
                TOUT* __restrict__ S,         // [NB, Cdst, TT]
                int O, int Cdst, int coff, int total) {
    int idx = blockIdx.x * blockDim.x + threadIdx.x;
    if (idx >= total) return;
    const int n = idx / O, o = idx - n * O;

    const float* __restrict__ y = Y + (size_t)idx * TT;
    TOUT* __restrict__ s = S + ((size_t)n * Cdst + coff + o) * TT;

    float u = 0.0f, v = 0.0f;
    s[0] = (TOUT)0.0f;                    // delay shift: t=0 is zero
    for (int t = 0; t < TT; ++t) {
        u = truncf(u * 0.75f)    + 64.0f * y[t];
        v = truncf(v * 0.96875f) + u;
        const float sp = (v >= 5120.0f) ? 1.0f : 0.0f;
        v *= (1.0f - sp);
        if (t + 1 < TT) s[t + 1] = (TOUT)sp;  // spike[TT-1] dropped by shift
    }
}

// ---------------------------------------------------------------------------
extern "C" void kernel_launch(void* const* d_in, const int* in_sizes, int n_in,
                              void* d_out, int out_size, void* d_ws, size_t ws_size,
                              hipStream_t stream) {
    (void)in_sizes; (void)n_in; (void)out_size; (void)ws_size;

    const float* tact = (const float*)d_in[0];  // [32, 156, 400]
    const float* vis  = (const float*)d_in[1];  // [32, 6300, 400]
    const float* Wt   = (const float*)d_in[2];  // [512, 156]
    const float* Wv   = (const float*)d_in[3];  // [512, 6300]
    const float* Wc   = (const float*)d_in[4];  // [256, 1024]
    float* out = (float*)d_out;                 // [32, 256, 400]

    const int NB = 32;
    const int KT = 156,  KTp = 160;
    const int KV = 6300, KVp = 6304;
    const int KC = 1024, KCp = 1024;
    const int M1 = 512, MC = 256;

    // workspace partition (256B aligned)
    char* ws = (char*)d_ws;
    size_t off = 0;
    auto take = [&](size_t bytes) -> char* {
        char* p = ws + off;
        off = (off + bytes + 255) & ~(size_t)255;
        return p;
    };
    half_t* Wt16 = (half_t*)take((size_t)M1 * KTp * sizeof(half_t));
    half_t* Wv16 = (half_t*)take((size_t)M1 * KVp * sizeof(half_t));
    half_t* Wc16 = (half_t*)take((size_t)MC * KCp * sizeof(half_t));
    float*  Yb   = (float*) take((size_t)NB * M1 * TT * sizeof(float));    // reused
    half_t* Scmb = (half_t*)take((size_t)NB * 1024 * TT * sizeof(half_t)); // spikes

    // 1) weight conversion (f32 -> f16, zero-padded K)
    cvt_pad_weights<<<(M1 * KTp + 255) / 256, 256, 0, stream>>>(Wt, Wt16, M1, KT, KTp);
    cvt_pad_weights<<<(M1 * KVp + 255) / 256, 256, 0, stream>>>(Wv, Wv16, M1, KV, KVp);
    cvt_pad_weights<<<(MC * KCp + 255) / 256, 256, 0, stream>>>(Wc, Wc16, MC, KC, KCp);

    // 2) tact branch: M=512 -> gridx = 512/(8*2*16) = 2
    dim3 g1(M1 / 256, TT / 80, NB);
    gemm_wmma<float, 2><<<g1, 256, 0, stream>>>(Wt16, tact, Yb, M1, KT, KTp);
    loihi_scan<half_t><<<(NB * M1 + 255) / 256, 256, 0, stream>>>(
        Yb, Scmb, M1, 1024, 0, NB * M1);

    // 3) vis branch (dominant GEMM)
    gemm_wmma<float, 2><<<g1, 256, 0, stream>>>(Wv16, vis, Yb, M1, KV, KVp);
    loihi_scan<half_t><<<(NB * M1 + 255) / 256, 256, 0, stream>>>(
        Yb, Scmb, M1, 1024, 512, NB * M1);

    // 4) combi stage: f16 spike B matrix straight from workspace
    dim3 g2(MC / 128, TT / 80, NB);
    gemm_wmma<half_t, 1><<<g2, 256, 0, stream>>>(Wc16, Scmb, Yb, MC, KC, KCp);
    loihi_scan<float><<<(NB * MC + 255) / 256, 256, 0, stream>>>(
        Yb, out, MC, MC, 0, NB * MC);
}
